// KernelMachine_29008209117207
// MI455X (gfx1250) — matrix operand order repaired and checked by reading, not compiled, not run on hardware
//
#include <hip/hip_runtime.h>
#include <hip/hip_bf16.h>
#include <stdint.h>

typedef __attribute__((ext_vector_type(16))) __bf16 v16bf;
typedef __attribute__((ext_vector_type(8)))  __bf16 v8bf;
typedef __attribute__((ext_vector_type(8)))  float  v8f;

#define NS      8192            // N_SAMPLES
#define NB      2048            // batch (inputs rows)
#define MT      (NS + NB)       // 10240 total rows
#define DIN     256
#define CHUNK   64
#define REST    192

union BF16x16 { v16bf v; v8bf h[2]; };

static __device__ __forceinline__ v8f wmma_bf16(v16bf a, v16bf b, v8f c) {
    return __builtin_amdgcn_wmma_f32_16x16x32_bf16(false, a, false, b, (short)0, c, false, false);
}

// A fragment (16x32, bf16, row-major source with K contiguous).
// Lane L: m = m0 + (L&15), half = L>>4; elems 0..7 -> K = k0+half*8+e, elems 8..15 -> K = k0+16+half*8+(e-8)
static __device__ __forceinline__ v16bf load_a_bf16(const __bf16* __restrict__ A, int pitch,
                                                    int m0, int k0, int lane) {
    int m = m0 + (lane & 15);
    int half = lane >> 4;
    const __bf16* p = A + (size_t)m * pitch + (k0 + half * 8);
    BF16x16 r;
    r.h[0] = *(const v8bf*)(p);
    r.h[1] = *(const v8bf*)(p + 16);
    return r.v;
}

// B fragment (32x16) from a transposed-stored operand BT[n][k] (K contiguous).
// Lane L: n = n0 + (L&15), half = L>>4; elems 0..7 -> K = k0+half*8+e, elems 8..15 -> K = k0+16+half*8+(e-8)
static __device__ __forceinline__ v16bf load_bt_bf16(const __bf16* __restrict__ BT, int pitch,
                                                     int k0, int n0, int lane) {
    int n = n0 + (lane & 15);
    int half = lane >> 4;
    const __bf16* p = BT + (size_t)n * pitch + (k0 + half * 8);
    BF16x16 r;
    r.h[0] = *(const v8bf*)(p);
    r.h[1] = *(const v8bf*)(p + 16);
    return r.v;
}

// A fragment from f32 source with in-register cvt to bf16 (used for ker @ cs)
static __device__ __forceinline__ v16bf load_a_f32(const float* __restrict__ A, size_t pitch,
                                                   int m0, int k0, int lane) {
    int m = m0 + (lane & 15);
    int half = lane >> 4;
    const float* p = A + (size_t)m * pitch + (k0 + half * 8);
    float4 a0 = *(const float4*)(p);
    float4 a1 = *(const float4*)(p + 4);
    float4 b0 = *(const float4*)(p + 16);
    float4 b1 = *(const float4*)(p + 20);
    v16bf r;
    r[0]=(__bf16)a0.x;  r[1]=(__bf16)a0.y;  r[2]=(__bf16)a0.z;  r[3]=(__bf16)a0.w;
    r[4]=(__bf16)a1.x;  r[5]=(__bf16)a1.y;  r[6]=(__bf16)a1.z;  r[7]=(__bf16)a1.w;
    r[8]=(__bf16)b0.x;  r[9]=(__bf16)b0.y;  r[10]=(__bf16)b0.z; r[11]=(__bf16)b0.w;
    r[12]=(__bf16)b1.x; r[13]=(__bf16)b1.y; r[14]=(__bf16)b1.z; r[15]=(__bf16)b1.w;
    return r;
}

// ---------------- conversion / setup kernels ----------------

__global__ __launch_bounds__(256) void conv_in_kernel(const float* __restrict__ data,
                                                      const float* __restrict__ inputs,
                                                      __bf16* __restrict__ xbin) {
    int t = blockIdx.x * 256 + threadIdx.x;           // t < MT*DIN, exact grid
    int r = t >> 8;
    float v = (r < NS) ? data[t] : inputs[t - NS * DIN];
    xbin[t] = (__bf16)v;
}

__global__ __launch_bounds__(256) void conv_w_kernel(const float* __restrict__ W,
                                                     __bf16* __restrict__ Wb) {
    int t = blockIdx.x * 256 + threadIdx.x;           // t < 65536
    Wb[t] = (__bf16)W[t];
}

__global__ __launch_bounds__(256) void reg_init_kernel(const float* __restrict__ W,
                                                       const float* __restrict__ b,
                                                       float* __restrict__ reg) {
    __shared__ float red[256];
    float s = 0.0f;
    for (int i = threadIdx.x; i < DIN * DIN; i += 256) { float w = W[i]; s += w * w; }
    if (threadIdx.x < DIN) { float bb = b[threadIdx.x]; s += bb * bb; }
    red[threadIdx.x] = s;
    __syncthreads();
    for (int st = 128; st > 0; st >>= 1) {
        if (threadIdx.x < st) red[threadIdx.x] += red[threadIdx.x + st];
        __syncthreads();
    }
    if (threadIdx.x == 0) reg[0] = red[0];
}

// per-stage: bf16 copy of the 64-col chunk of x + row norms
__global__ __launch_bounds__(256) void stage_prep_kernel(const float* __restrict__ x,
                                                         __bf16* __restrict__ xs,
                                                         float* __restrict__ uu, int stage) {
    int m = blockIdx.x * 256 + threadIdx.x;           // m < MT, exact grid
    const float* row = x + (size_t)m * DIN + stage * CHUNK;
    float s = 0.0f;
    #pragma unroll
    for (int j = 0; j < CHUNK; j += 4) {
        float4 v = *(const float4*)(row + j);
        s += v.x * v.x + v.y * v.y + v.z * v.z + v.w * v.w;
        __bf16* o = xs + (size_t)m * CHUNK + j;
        o[0] = (__bf16)v.x; o[1] = (__bf16)v.y; o[2] = (__bf16)v.z; o[3] = (__bf16)v.w;
    }
    uu[m] = s;
}

// csT[j][k] = bf16(c_ss[k][64*stage + j]) : transposed bf16 copy for B fragments
__global__ __launch_bounds__(256) void cst_prep_kernel(const float* __restrict__ c_ss,
                                                       __bf16* __restrict__ csT, int stage) {
    int t = blockIdx.x * 256 + threadIdx.x;           // t < 64*8192, exact grid
    int k = t & (NS - 1);
    int j = t >> 13;
    csT[(size_t)j * NS + k] = (__bf16)c_ss[(size_t)k * REST + stage * CHUNK + j];
}

// ---------------- x = [data;inputs] @ W^T + b  (M=10240, N=256, K=256) ----------------

__global__ __launch_bounds__(256) void gemm_x_kernel(const __bf16* __restrict__ xbin,
                                                     const __bf16* __restrict__ Wb,
                                                     const float* __restrict__ bias,
                                                     float* __restrict__ x) {
    int lane = threadIdx.x & 31;
    int wid  = threadIdx.x >> 5;
    int wm = wid & 1, wn = wid >> 1;
    int m_base = blockIdx.x * 128 + wm * 64;
    int n_base = wn * 64;

    v8f acc[4][4] = {};
    for (int k0 = 0; k0 < DIN; k0 += 32) {
        v16bf a[4], bt[4];
        #pragma unroll
        for (int im = 0; im < 4; ++im) a[im]  = load_a_bf16(xbin, DIN, m_base + 16 * im, k0, lane);
        #pragma unroll
        for (int in = 0; in < 4; ++in) bt[in] = load_bt_bf16(Wb, DIN, k0, n_base + 16 * in, lane);
        #pragma unroll
        for (int im = 0; im < 4; ++im)
            #pragma unroll
            for (int in = 0; in < 4; ++in)
                acc[im][in] = wmma_bf16(a[im], bt[in], acc[im][in]);
    }

    float bb[4];
    #pragma unroll
    for (int in = 0; in < 4; ++in) bb[in] = bias[n_base + 16 * in + (lane & 15)];
    #pragma unroll
    for (int im = 0; im < 4; ++im)
        #pragma unroll
        for (int in = 0; in < 4; ++in) {
            int n = n_base + 16 * in + (lane & 15);
            #pragma unroll
            for (int r = 0; r < 8; ++r) {
                int m = m_base + 16 * im + (lane >> 4) * 8 + r;
                x[(size_t)m * DIN + n] = acc[im][in][r] + bb[in];
            }
        }
}

// ---------------- ker (+)= exp(xs.xs^T - uu/2 - vv/2)  (M=10240, N=8192, K=64) ----------------

__global__ __launch_bounds__(256) void k1_kernel(const __bf16* __restrict__ xs,
                                                 const float* __restrict__ uu,
                                                 float* __restrict__ ker, int accumulate) {
    int lane = threadIdx.x & 31;
    int wid  = threadIdx.x >> 5;
    int wm = wid & 1, wn = wid >> 1;
    int m_base = blockIdx.y * 128 + wm * 64;
    int n_base = blockIdx.x * 256 + wn * 64;

    v8f acc[4][4] = {};
    #pragma unroll
    for (int k0 = 0; k0 < CHUNK; k0 += 32) {
        v16bf a[4], bt[4];
        #pragma unroll
        for (int im = 0; im < 4; ++im) a[im]  = load_a_bf16(xs, CHUNK, m_base + 16 * im, k0, lane);
        #pragma unroll
        for (int in = 0; in < 4; ++in) bt[in] = load_bt_bf16(xs, CHUNK, k0, n_base + 16 * in, lane);
        #pragma unroll
        for (int im = 0; im < 4; ++im)
            #pragma unroll
            for (int in = 0; in < 4; ++in)
                acc[im][in] = wmma_bf16(a[im], bt[in], acc[im][in]);
    }

    float un[4];
    #pragma unroll
    for (int in = 0; in < 4; ++in) un[in] = uu[n_base + 16 * in + (lane & 15)];

    #pragma unroll
    for (int im = 0; im < 4; ++im) {
        float um[8];
        #pragma unroll
        for (int r = 0; r < 8; ++r) um[r] = uu[m_base + 16 * im + (lane >> 4) * 8 + r];
        #pragma unroll
        for (int in = 0; in < 4; ++in) {
            int n = n_base + 16 * in + (lane & 15);
            #pragma unroll
            for (int r = 0; r < 8; ++r) {
                int m = m_base + 16 * im + (lane >> 4) * 8 + r;
                float e = __expf(acc[im][in][r] - 0.5f * (um[r] + un[in]));
                size_t idx = (size_t)m * NS + n;
                if (accumulate) ker[idx] += e; else ker[idx] = e;
            }
        }
    }
}

// ---------------- val = ker @ cs ; x[:,64*(s+1)] += val ; reg partial  (M=10240, N=64, K=8192) ----

__global__ __launch_bounds__(256) void k2_kernel(const float* __restrict__ ker,
                                                 const __bf16* __restrict__ csT,
                                                 const float* __restrict__ c_ss,
                                                 float* __restrict__ x,
                                                 float* __restrict__ regp, int stage) {
    int lane = threadIdx.x & 31;
    int wid  = threadIdx.x >> 5;
    int m_base = blockIdx.x * 128 + wid * 16;

    v8f acc[4] = {};
    for (int k0 = 0; k0 < NS; k0 += 32) {
        v16bf a = load_a_f32(ker, NS, m_base, k0, lane);
        #pragma unroll
        for (int in = 0; in < 4; ++in) {
            v16bf bt = load_bt_bf16(csT, NS, k0, 16 * in, lane);
            acc[in] = wmma_bf16(a, bt, acc[in]);
        }
    }

    float p = 0.0f;
    #pragma unroll
    for (int in = 0; in < 4; ++in) {
        int nl = 16 * in + (lane & 15);
        #pragma unroll
        for (int r = 0; r < 8; ++r) {
            int m = m_base + (lane >> 4) * 8 + r;
            float v = acc[in][r];
            x[(size_t)m * DIN + (stage + 1) * CHUNK + nl] += v;
            if (m < NS) p += c_ss[(size_t)m * REST + stage * CHUNK + nl] * v;
        }
    }

    __shared__ float red[256];
    red[threadIdx.x] = p;
    __syncthreads();
    for (int st = 128; st > 0; st >>= 1) {
        if (threadIdx.x < st) red[threadIdx.x] += red[threadIdx.x + st];
        __syncthreads();
    }
    if (threadIdx.x == 0) regp[stage * 80 + blockIdx.x] = red[0];
}

// ---------------- final gather ----------------

__global__ __launch_bounds__(256) void final_kernel(const float* __restrict__ x,
                                                    const float* __restrict__ reg,
                                                    const float* __restrict__ regp,
                                                    float* __restrict__ out) {
    int t = blockIdx.x * 256 + threadIdx.x;           // t < 2048*64, exact grid
    int r = t >> 6, j = t & 63;
    out[t] = x[(size_t)(NS + r) * DIN + REST + j];
    if (t == 0) {
        float s = reg[0];
        for (int i = 0; i < 240; ++i) s += regp[i];
        out[NB * CHUNK] = s;
    }
}

// ---------------- host launch ----------------

extern "C" void kernel_launch(void* const* d_in, const int* in_sizes, int n_in,
                              void* d_out, int out_size, void* d_ws, size_t ws_size,
                              hipStream_t stream) {
    const float* d_data   = (const float*)d_in[0];
    const float* d_inputs = (const float*)d_in[1];
    const float* d_W      = (const float*)d_in[2];
    const float* d_b      = (const float*)d_in[3];
    const float* d_css    = (const float*)d_in[4];
    float* out = (float*)d_out;

    char* ws = (char*)d_ws;
    const size_t OFF_KER  = 0;                                   // 10240*8192*4 = 335544320
    const size_t OFF_X    = OFF_KER  + (size_t)MT * NS * 4;      // 10240*256*4
    const size_t OFF_XBIN = OFF_X    + (size_t)MT * DIN * 4;     // 10240*256*2
    const size_t OFF_WB   = OFF_XBIN + (size_t)MT * DIN * 2;     // 256*256*2
    const size_t OFF_XS   = OFF_WB   + (size_t)DIN * DIN * 2;    // 10240*64*2
    const size_t OFF_UU   = OFF_XS   + (size_t)MT * CHUNK * 2;   // 10240*4
    const size_t OFF_CST  = OFF_UU   + (size_t)MT * 4;           // 64*8192*2
    const size_t OFF_REG  = OFF_CST  + (size_t)CHUNK * NS * 2;   // 4
    const size_t OFF_REGP = OFF_REG  + 256;                      // 240*4

    float*  ker  = (float*)(ws + OFF_KER);
    float*  x    = (float*)(ws + OFF_X);
    __bf16* xbin = (__bf16*)(ws + OFF_XBIN);
    __bf16* Wb   = (__bf16*)(ws + OFF_WB);
    __bf16* xs   = (__bf16*)(ws + OFF_XS);
    float*  uu   = (float*)(ws + OFF_UU);
    __bf16* csT  = (__bf16*)(ws + OFF_CST);
    float*  reg  = (float*)(ws + OFF_REG);
    float*  regp = (float*)(ws + OFF_REGP);

    conv_in_kernel<<<MT, 256, 0, stream>>>(d_data, d_inputs, xbin);
    conv_w_kernel<<<DIN, 256, 0, stream>>>(d_W, Wb);
    reg_init_kernel<<<1, 256, 0, stream>>>(d_W, d_b, reg);
    gemm_x_kernel<<<MT / 128, 256, 0, stream>>>(xbin, Wb, d_b, x);

    for (int i = 0; i < 3; ++i) {
        stage_prep_kernel<<<MT / 256, 256, 0, stream>>>(x, xs, uu, i);
        cst_prep_kernel<<<(CHUNK * NS) / 256, 256, 0, stream>>>(d_css, csT, i);
        k1_kernel<<<dim3(NS / 256, MT / 128), 256, 0, stream>>>(xs, uu, ker, i > 0);
        k2_kernel<<<MT / 128, 256, 0, stream>>>(ker, csT, d_css, x, regp, i);
    }

    final_kernel<<<(NB * CHUNK) / 256, 256, 0, stream>>>(x, reg, regp, out);
}
